// SimpleSparseCNN_18829136626385
// MI455X (gfx1250) — matrix operand order repaired and checked by reading, not compiled, not run on hardware
//
#include <hip/hip_runtime.h>
#include <hip/hip_bf16.h>
#include <math.h>

// ---------------------------------------------------------------------------
// SimpleSparseCNN fused: per-channel 2x2 conv + tanh (+ red mask) feeding a
// tall-skinny GEMM (M=128 batch, N=64, K=65025) per color, via
// v_wmma_f32_16x16x32_bf16 on gfx1250 (wave32).
//
// Roofline: x (100MB) + W (50MB) ~= 150MB -> ~6.5us @ 23.3 TB/s. Each WG
// covers ALL 128 batches x ALL 64 outputs for a K-slice of rows, so x and W
// stream through ~once (the 192MB L2 holds nearly the whole working set).
// tanh uses the gfx1250 hardware V_TANH_F32 trans op so the 25M activation
// tanhs stay off the critical VALU path.
// ---------------------------------------------------------------------------

typedef float  v8f   __attribute__((ext_vector_type(8)));
typedef __bf16 v4bf  __attribute__((ext_vector_type(4)));
typedef __bf16 v8bf  __attribute__((ext_vector_type(8)));
typedef __bf16 v16bf __attribute__((ext_vector_type(16)));
typedef float  f4u   __attribute__((ext_vector_type(4), aligned(4)));  // unaligned float4
typedef float  f4a   __attribute__((ext_vector_type(4)));              // 16B-aligned float4

#define BATCH  128
#define HW     256
#define CSIZE  255              // conv output edge
#define KTOT   (255 * 255)      // 65025
#define OUT3   64

#if __has_builtin(__builtin_amdgcn_tanhf)
__device__ __forceinline__ float fast_tanh(float v) { return __builtin_amdgcn_tanhf(v); }
#else
__device__ __forceinline__ float fast_tanh(float v) { return tanhf(v); }
#endif

__global__ __launch_bounds__(256)
void conv_wmma_kernel(const float* __restrict__ x,
                      const float* __restrict__ w_red,
                      const float* __restrict__ w_green,
                      const float* __restrict__ w_blue,
                      const float* __restrict__ Wr,
                      const float* __restrict__ Wg,
                      const float* __restrict__ Wb,
                      float* __restrict__ partial,
                      int rows_per_wg)
{
    // act_lds[b][j]: bf16 activation, batch b, column j of current row (j=255 pad=0)
    // w_lds[n][j]:   bf16 W[n, r*255+j]                         (j=255 pad=0)
    __shared__ __align__(32) __bf16 act_lds[BATCH * 256];   // 64 KB
    __shared__ __align__(32) __bf16 w_lds[OUT3 * 256];      // 32 KB

    const int c  = blockIdx.y;                   // color 0..2
    const int rb = blockIdx.x;                   // row block within color
    const int r0 = rb * rows_per_wg;
    const int blk = c * gridDim.x + rb;

    const float* Wc = (c == 0) ? Wr : (c == 1) ? Wg : Wb;
    const float* wc = (c == 0) ? w_red : (c == 1) ? w_green : w_blue;
    const float k00 = wc[0], k01 = wc[1], k10 = wc[2], k11 = wc[3];

    const int tid  = threadIdx.x;
    const int lane = tid & 31;
    const int wv   = tid >> 5;       // wave 0..7 -> M tile (batches 16*wv..+15)
    const int half = lane >> 4;      // K-half selector per WMMA 16-bit layout
    const int nloc = lane & 15;      // M row (A) / N column (B) within tile

    v8f acc[4] = {};                 // 4 N-tiles of 16x16 f32 accumulators

    for (int rr = 0; rr < rows_per_wg; ++rr) {
        const int r = r0 + rr;

        // ---- Phase A1: stage W row slice (fp32 -> bf16) into LDS ----------
        // 64 n * 64 groups-of-4 = 4096 -> 16 iters/thread
        for (int it = tid; it < OUT3 * 64; it += 256) {
            const int n = it >> 6, g = it & 63;
            const float* wp = Wc + (size_t)n * KTOT + (size_t)r * CSIZE + 4 * g;
            float w0, w1, w2, w3;
            if (g < 63) {                           // rows are 255 long: last group pads
                f4u t = *(const f4u*)wp;            // unaligned-ok b128 load
                w0 = t.x; w1 = t.y; w2 = t.z; w3 = t.w;
            } else {
                w0 = wp[0]; w1 = wp[1]; w2 = wp[2]; w3 = 0.0f;
            }
            v4bf o = {(__bf16)w0, (__bf16)w1, (__bf16)w2, (__bf16)w3};
            *(v4bf*)(w_lds + (n << 8) + 4 * g) = o;   // ds_store_b64
            if ((g & 7) == 0 && (rr + 1) < rows_per_wg)  // prefetch next row's W
                __builtin_prefetch(&Wc[(size_t)n * KTOT + (size_t)(r + 1) * CSIZE + 4 * g], 0, 1);
        }

        // ---- Phase A2: conv 2x2 + tanh (+ red mask), fp32 -> bf16 in LDS --
        // 128 b * 64 groups-of-4 = 8192 -> 32 iters/thread, float4 loads
        for (int it = tid; it < BATCH * 64; it += 256) {
            const int b = it >> 6, g = it & 63;
            const float* xb = x + (((size_t)b * 3 + c) * HW + r) * HW + 4 * g;
            f4a a0 = *(const f4a*)xb;            // row r   (16B aligned)
            f4a a1 = *(const f4a*)(xb + HW);     // row r+1 (16B aligned)
            float n0 = 0.0f, n1 = 0.0f;
            if (g < 63) { n0 = xb[4]; n1 = xb[HW + 4]; }
            const float e0[5] = {a0.x, a0.y, a0.z, a0.w, n0};
            const float e1[5] = {a1.x, a1.y, a1.z, a1.w, n1};
            v4bf o;
#pragma unroll
            for (int u = 0; u < 4; ++u) {
                const int j = 4 * g + u;
                float v = 0.0f;
                if (j < CSIZE) {
                    v = fast_tanh(k00 * e0[u] + k01 * e0[u + 1] +
                                  k10 * e1[u] + k11 * e1[u + 1]);
                    if (c == 0 && e0[u] == 0.0f) v = 0.0f;   // red sparsity mask
                }
                o[u] = (__bf16)v;
            }
            *(v4bf*)(act_lds + (b << 8) + 4 * g) = o;        // ds_store_b64
        }
        __syncthreads();

        // ---- Phase B: 8 K-steps of 32 -> v_wmma_f32_16x16x32_bf16 ---------
#pragma unroll
        for (int j0 = 0; j0 < 256; j0 += 32) {
            // A fragment (16x32): lane m=nloc; lanes<16 K=0..15, lanes>=16 K=16..31
            const __bf16* ap = act_lds + ((16 * wv + nloc) << 8) + j0 + half * 16;
            v8bf alo = *(const v8bf*)ap;          // ds_load_b128
            v8bf ahi = *(const v8bf*)(ap + 8);    // ds_load_b128
            v16bf afrag = __builtin_shufflevector(alo, ahi,
                0, 1, 2, 3, 4, 5, 6, 7, 8, 9, 10, 11, 12, 13, 14, 15);
#pragma unroll
            for (int nt = 0; nt < 4; ++nt) {
                // B fragment (32x16): lane col n=nloc; lanes<16 K=0..15, lanes>=16 K=16..31
                const __bf16* bp = w_lds + ((nt * 16 + nloc) << 8) + j0 + half * 16;
                v16bf bfrag = *(const v16bf*)bp;  // two ds_load_b128
                acc[nt] = __builtin_amdgcn_wmma_f32_16x16x32_bf16(
                    false, afrag, false, bfrag, (short)0, acc[nt], false, false);
            }
        }
        __syncthreads();
    }

    // ---- write partial C tiles: partial[blk][m][n] ------------------------
    // C layout: VGPR v, lanes<16 -> M=v, N=lane ; lanes>=16 -> M=8+v, N=lane-16
#pragma unroll
    for (int nt = 0; nt < 4; ++nt) {
#pragma unroll
        for (int v = 0; v < 8; ++v) {
            const int m = 16 * wv + half * 8 + v;
            const int n = nt * 16 + nloc;
            partial[((size_t)blk * BATCH + m) * OUT3 + n] = acc[nt][v];
        }
    }
}

__global__ __launch_bounds__(256)
void reduce_tanh_kernel(const float* __restrict__ partial,
                        const float* __restrict__ br,
                        const float* __restrict__ bg,
                        const float* __restrict__ bb,
                        float* __restrict__ out,
                        int wgs_per_clr)
{
    const int t   = blockIdx.x * 256 + threadIdx.x;   // < 128*192
    const int b   = t / 192;
    const int col = t % 192;
    const int c   = col >> 6;
    const int n   = col & 63;
    const float* bias = (c == 0) ? br : (c == 1) ? bg : bb;

    float s = bias[n];
    const float* p = partial + ((size_t)(c * wgs_per_clr) * BATCH + b) * OUT3 + n;
    for (int rbk = 0; rbk < wgs_per_clr; ++rbk)
        s += p[(size_t)rbk * BATCH * OUT3];
    out[t] = fast_tanh(s);
}

extern "C" void kernel_launch(void* const* d_in, const int* in_sizes, int n_in,
                              void* d_out, int out_size, void* d_ws, size_t ws_size,
                              hipStream_t stream)
{
    const float* x      = (const float*)d_in[0];
    const float* w_red  = (const float*)d_in[1];
    const float* w_grn  = (const float*)d_in[2];
    const float* w_blu  = (const float*)d_in[3];
    const float* Wr     = (const float*)d_in[4];
    const float* br     = (const float*)d_in[5];
    const float* Wg     = (const float*)d_in[6];
    const float* bg     = (const float*)d_in[7];
    const float* Wb     = (const float*)d_in[8];
    const float* bb     = (const float*)d_in[9];
    float* out     = (float*)d_out;
    float* partial = (float*)d_ws;

    // Pick finest K-slicing (most parallelism) whose partial buffer fits ws.
    // rows must divide 255. rows=3 -> 255 WGs, 8.4MB partials (preferred).
    static const int cand[] = {3, 5, 15, 17, 51, 85, 255};
    int rows = 255, wgs = 1;
    for (int i = 0; i < 7; ++i) {
        const int w = 255 / cand[i];
        const size_t need = (size_t)3 * w * BATCH * OUT3 * sizeof(float);
        if (need <= ws_size) { rows = cand[i]; wgs = w; break; }
    }

    conv_wmma_kernel<<<dim3(wgs, 3), dim3(256), 0, stream>>>(
        x, w_red, w_grn, w_blu, Wr, Wg, Wb, partial, rows);

    reduce_tanh_kernel<<<dim3((BATCH * 192) / 256), dim3(256), 0, stream>>>(
        partial, br, bg, bb, out, wgs);
}
